// DP_29652454211852
// MI455X (gfx1250) — matrix operand order repaired and checked by reading, not compiled, not run on hardware
//
#include <hip/hip_runtime.h>

typedef __attribute__((ext_vector_type(16))) _Float16 v16h;
typedef __attribute__((ext_vector_type(8)))  float    v8f;

#define NTYPES 2
#define MAXNEI 128
#define NPER   256
#define E0 25
#define E1 50
#define E2 100
#define FIT_IN 1600
#define FH 240

// ---- workspace layout (units: _Float16 elements) ----
#define OFF_EW1T 0                       // emb W1^T  [4][64][32]
#define SZ_EW1T  (4*64*32)
#define OFF_EW2T (OFF_EW1T + SZ_EW1T)    // emb W2^T  [4][112][64]
#define SZ_EW2T  (4*112*64)
#define OFF_FW0T (OFF_EW2T + SZ_EW2T)    // fit W0^T  [2][240][1600]
#define SZ_FW0T  (2*240*1600)
#define OFF_FW1T (OFF_FW0T + SZ_FW0T)    // fit W1^T  [2][240][256]
#define SZ_FWxT  (2*240*256)
#define OFF_FW2T (OFF_FW1T + SZ_FWxT)    // fit W2^T  [2][240][256]
#define OFF_DR   (OFF_FW2T + SZ_FWxT)    // DR (d)    [2][1024][1600]
// total ~8.5 MB of workspace

#define WMMA(a,b,c) __builtin_amdgcn_wmma_f32_16x16x32_f16(false,(a),false,(b),(short)0,(c),false,false)

// Branch-free tanh: prefer the gfx1250 hardware V_TANH_F32; otherwise use
// exp/rcp hardware transcendentals (no EXEC divergence, co-executes with WMMA).
__device__ __forceinline__ float tanh_fast(float x) {
#if __has_builtin(__builtin_amdgcn_tanhf)
  return __builtin_amdgcn_tanhf(x);
#else
  float e = __expf(2.f * x);
  return 1.f - 2.f * __builtin_amdgcn_rcpf(e + 1.f);
#endif
}

// A-fragment (16x32, 16-bit): lane holds M=lane%16; K pattern per ISA 7.12.2:
// VGPR v<4: K=2v,2v+1 (+8 for lanes>=16); v>=4: K=16+2(v-4) (+8).
__device__ __forceinline__ v16h frag_a(const _Float16* p, int ld, int lane) {
  int row = lane & 15;
  int kb  = (lane & 16) ? 8 : 0;
  const _Float16* rp = p + row * ld + kb;
  union { v16h v; unsigned int u[8]; } f;
#pragma unroll
  for (int i = 0; i < 8; ++i) {
    int k = (i < 4) ? (2 * i) : (16 + 2 * (i - 4));
    f.u[i] = *(const unsigned int*)(rp + k);
  }
  return f.v;
}

// B-fragment (32x16, 16-bit), source stored transposed as BT[n][k]:
// lane holds N=lane%16; K pattern matches frag_a: v<4: K=2v,2v+1 (+8 for lanes>=16); v>=4: K=16+2(v-4) (+8).
__device__ __forceinline__ v16h frag_b(const _Float16* p, int ld, int lane) {
  int row = lane & 15;
  int kb  = (lane & 16) ? 8 : 0;
  const _Float16* rp = p + row * ld + kb;
  union { v16h v; unsigned int u[8]; } f;
#pragma unroll
  for (int i = 0; i < 8; ++i) f.u[i] = *(const unsigned int*)(rp + ((i < 4) ? (2 * i) : (16 + 2 * (i - 4))));
  return f.v;
}

// ---------------- prep: f32 weights -> padded transposed f16 ----------------
__global__ void prep_kernel(const float* __restrict__ eW1, const float* __restrict__ eW2,
                            const float* __restrict__ fW0, const float* __restrict__ fW1,
                            const float* __restrict__ fW2, _Float16* __restrict__ ws) {
  int tid = blockIdx.x * blockDim.x + threadIdx.x;
  int stride = gridDim.x * blockDim.x;
  for (int u = tid; u < SZ_EW1T; u += stride) {              // (4,25,50) -> [e][n<64][k<32]
    int e = u / (64 * 32), r = u % (64 * 32), n = r / 32, k = r % 32;
    float v = (n < E1 && k < E0) ? eW1[(e * E0 + k) * E1 + n] : 0.f;
    ws[OFF_EW1T + u] = (_Float16)v;
  }
  for (int u = tid; u < SZ_EW2T; u += stride) {              // (4,50,100) -> [e][g<112][k<64]
    int e = u / (112 * 64), r = u % (112 * 64), g = r / 64, k = r % 64;
    float v = (g < E2 && k < E1) ? eW2[(e * E1 + k) * E2 + g] : 0.f;
    ws[OFF_EW2T + u] = (_Float16)v;
  }
  for (int u = tid; u < SZ_FW0T; u += stride) {              // (2,1600,240) -> [i][n][k]
    int i = u / (240 * 1600), r = u % (240 * 1600), n = r / 1600, k = r % 1600;
    ws[OFF_FW0T + u] = (_Float16)fW0[(i * FIT_IN + k) * FH + n];
  }
  for (int u = tid; u < SZ_FWxT; u += stride) {              // (2,240,240) -> [i][n][k<256]
    int i = u / (240 * 256), r = u % (240 * 256), n = r / 256, k = r % 256;
    float v1 = (k < FH) ? fW1[(i * FH + k) * FH + n] : 0.f;
    float v2 = (k < FH) ? fW2[(i * FH + k) * FH + n] : 0.f;
    ws[OFF_FW1T + u] = (_Float16)v1;
    ws[OFF_FW2T + u] = (_Float16)v2;
  }
}

// ---------------- embedding + descriptor: one WG (128 thr) per atom ----------------
__global__ __launch_bounds__(128) void embed_kernel(
    const float* __restrict__ Ri,
    const float* __restrict__ eW0, const float* __restrict__ eb0,
    const float* __restrict__ eb1, const float* __restrict__ eb2,
    const _Float16* __restrict__ ws, _Float16* __restrict__ DRout) {
  __shared__ __align__(16) _Float16 blkT[16][128];   // A of tmp-GEMM (rows 4..15 zero)
  __shared__ __align__(16) _Float16 h0[128][32];     // cols 25..31 zero (K-pad)
  __shared__ __align__(16) _Float16 h1[128][64];     // cols 50..63 zero (K-pad)
  __shared__ __align__(16) _Float16 GT[112][128];    // G transposed = B of tmp-GEMM
  __shared__ float xyzf[4][112];

  const int t = threadIdx.x;
  const int lane = t & 31;
  const int wave = t >> 5;
  const int atom = blockIdx.x;         // 0..2047
  const int b = atom >> 9;
  const int a = atom & 511;            // i*256 + n
  const int ctype = a >> 8;
  const int nloc  = a & 255;

  const _Float16* eW1T = ws + OFF_EW1T;
  const _Float16* eW2T = ws + OFF_EW2T;

  for (int u = t; u < 16 * 128; u += 128) ((_Float16*)blkT)[u] = (_Float16)0.f;
  for (int u = t; u < 128 * 32; u += 128) ((_Float16*)h0)[u]   = (_Float16)0.f;
  for (int u = t; u < 128 * 64; u += 128) ((_Float16*)h1)[u]   = (_Float16)0.f;
  __syncthreads();

  v8f accT[2];
#pragma unroll
  for (int s = 0; s < 2; ++s)
#pragma unroll
    for (int r = 0; r < 8; ++r) accT[s][r] = 0.f;

  for (int j = 0; j < NTYPES; ++j) {
    const int e = ctype * NTYPES + j;

    { // load neighbor row m=t, build blkT and h0
      const int m = t;
      const float* rp = Ri + ((((size_t)b * 512 + a) * 256) + (size_t)j * 128 + m) * 4;
      float c0 = rp[0], c1 = rp[1], c2 = rp[2], c3 = rp[3];
      blkT[0][m] = (_Float16)c0; blkT[1][m] = (_Float16)c1;
      blkT[2][m] = (_Float16)c2; blkT[3][m] = (_Float16)c3;
#pragma unroll
      for (int kk = 0; kk < E0; ++kk)
        h0[m][kk] = (_Float16)tanh_fast(c0 * eW0[e * E0 + kk] + eb0[e * E0 + kk]);
    }
    __syncthreads();

    // GEMM1: h1 = tanh(h0 @ W1 + b1) + [h0,h0]   (M=128,K=32,N=64)
#pragma unroll
    for (int mi = 0; mi < 2; ++mi) {
      int mt = 2 * wave + mi;
      v16h afr = frag_a(&h0[mt * 16][0], 32, lane);
#pragma unroll
      for (int nt = 0; nt < 4; ++nt) {
        v16h bfr = frag_b(eW1T + (e * 64 + nt * 16) * 32, 32, lane);
        v8f acc = {0.f,0.f,0.f,0.f,0.f,0.f,0.f,0.f};
        acc = WMMA(afr, bfr, acc);
        int n = nt * 16 + (lane & 15);
        if (n < E1) {
          int mb = mt * 16 + ((lane & 16) ? 8 : 0);
#pragma unroll
          for (int r = 0; r < 8; ++r) {
            int m = mb + r;
            h1[m][n] = (_Float16)(tanh_fast(acc[r] + eb1[e * E1 + n]) + (float)h0[m][n % E0]);
          }
        }
      }
    }
    __syncthreads();

    // GEMM2: G = tanh(h1 @ W2 + b2) + [h1,h1]  (M=128,K=64,N=112) -> GT transposed
#pragma unroll
    for (int ni = 0; ni < 2; ++ni) {
      int nt = wave + 4 * ni;
      if (nt >= 7) continue;            // wave-uniform
#pragma unroll
      for (int mt = 0; mt < 8; ++mt) {
        v8f acc = {0.f,0.f,0.f,0.f,0.f,0.f,0.f,0.f};
#pragma unroll
        for (int kt = 0; kt < 2; ++kt) {
          v16h afr = frag_a(&h1[mt * 16][kt * 32], 64, lane);
          v16h bfr = frag_b(eW2T + (e * 112 + nt * 16) * 64 + kt * 32, 64, lane);
          acc = WMMA(afr, bfr, acc);
        }
        int g  = nt * 16 + (lane & 15);
        int mb = mt * 16 + ((lane & 16) ? 8 : 0);
#pragma unroll
        for (int r = 0; r < 8; ++r) {
          int m = mb + r;
          float v = 0.f;
          if (g < E2) v = tanh_fast(acc[r] + eb2[e * E2 + g]) + (float)h1[m][g % E1];
          GT[g][m] = (_Float16)v;
        }
      }
    }
    __syncthreads();

    // tmp GEMM: blkT(16x128) @ G(128x112), accumulated over j
#pragma unroll
    for (int s = 0; s < 2; ++s) {
      int nt = 2 * wave + s;
      if (nt >= 7) continue;            // wave-uniform
#pragma unroll
      for (int kt = 0; kt < 4; ++kt) {
        v16h afr = frag_a(&blkT[0][kt * 32], 128, lane);
        v16h bfr = frag_b(&GT[nt * 16][kt * 32], 128, lane);
        accT[s] = WMMA(afr, bfr, accT[s]);
      }
    }
    __syncthreads();                     // protect LDS before next j overwrites
  }

  // xyz = tmp / (MAXNEI*NTYPES)
  if (lane < 16) {
#pragma unroll
    for (int s = 0; s < 2; ++s) {
      int nt = 2 * wave + s;
      if (nt < 7) {
        int g = nt * 16 + lane;
#pragma unroll
        for (int r = 0; r < 4; ++r) xyzf[r][g] = accT[s][r] * (1.f / 256.f);
      }
    }
  }
  __syncthreads();

  // DR[g][h] = sum_c xyz[c][g]*xyz[c][h], flat f = g*16+h
  _Float16* drp = DRout + ((size_t)ctype * 1024 + (size_t)b * 256 + nloc) * FIT_IN;
  for (int f = t; f < FIT_IN; f += 128) {
    int g = f >> 4, h = f & 15;
    float v = 0.f;
#pragma unroll
    for (int c = 0; c < 4; ++c) v += xyzf[c][g] * xyzf[c][h];
    drp[f] = (_Float16)v;
  }
}

// ---------------- fitting net: one WG per 16-row M-tile ----------------
__global__ __launch_bounds__(128) void fit_kernel(
    const _Float16* __restrict__ ws,
    const float* __restrict__ fb0, const float* __restrict__ fb1,
    const float* __restrict__ fb2, const float* __restrict__ fW3,
    const float* __restrict__ fb3, float* __restrict__ out) {
  __shared__ __align__(16) _Float16 dA[16][1600];
  __shared__ __align__(16) _Float16 hA[16][256];   // cols 240..255 zero (K-pad)
  __shared__ __align__(16) _Float16 hB[16][256];

  const int t = threadIdx.x;
  const int lane = t & 31;
  const int wave = t >> 5;
  const int wg = blockIdx.x;           // 0..127
  const int ctype = wg >> 6;
  const int mtile = wg & 63;

  const _Float16* fW0T = ws + OFF_FW0T + (size_t)ctype * 240 * 1600;
  const _Float16* fW1T = ws + OFF_FW1T + (size_t)ctype * 240 * 256;
  const _Float16* fW2T = ws + OFF_FW2T + (size_t)ctype * 240 * 256;
  const _Float16* DRsrc = ws + OFF_DR + ((size_t)ctype * 1024 + (size_t)mtile * 16) * FIT_IN;

  { // stage A-tile (16x1600 halves, contiguous) into LDS
    const uint4* s4 = (const uint4*)DRsrc;
    uint4* d4 = (uint4*)&dA[0][0];
    for (int u = t; u < 3200; u += 128) d4[u] = s4[u];
  }
  for (int u = t; u < 16 * 256; u += 128) {
    ((_Float16*)hA)[u] = (_Float16)0.f;
    ((_Float16*)hB)[u] = (_Float16)0.f;
  }
  __syncthreads();

  // GEMM0: hA = tanh(d @ W0 + b0)   K=1600 (50 tiles), N=240 (15 tiles)
#pragma unroll
  for (int ni = 0; ni < 4; ++ni) {
    int nt = wave + 4 * ni;
    if (nt >= 15) continue;            // wave-uniform
    v8f acc = {0.f,0.f,0.f,0.f,0.f,0.f,0.f,0.f};
    for (int kt = 0; kt < 50; ++kt) {
      v16h afr = frag_a(&dA[0][kt * 32], 1600, lane);
      v16h bfr = frag_b(fW0T + (size_t)(nt * 16) * 1600 + kt * 32, 1600, lane);
      acc = WMMA(afr, bfr, acc);
    }
    int n  = nt * 16 + (lane & 15);
    int mb = (lane & 16) ? 8 : 0;
#pragma unroll
    for (int r = 0; r < 8; ++r)
      hA[mb + r][n] = (_Float16)tanh_fast(acc[r] + fb0[ctype * FH + n]);
  }
  __syncthreads();

  // GEMM1: hB = tanh(hA @ W1 + b1) + hA
#pragma unroll
  for (int ni = 0; ni < 4; ++ni) {
    int nt = wave + 4 * ni;
    if (nt >= 15) continue;
    v8f acc = {0.f,0.f,0.f,0.f,0.f,0.f,0.f,0.f};
#pragma unroll
    for (int kt = 0; kt < 8; ++kt) {
      v16h afr = frag_a(&hA[0][kt * 32], 256, lane);
      v16h bfr = frag_b(fW1T + (size_t)(nt * 16) * 256 + kt * 32, 256, lane);
      acc = WMMA(afr, bfr, acc);
    }
    int n  = nt * 16 + (lane & 15);
    int mb = (lane & 16) ? 8 : 0;
#pragma unroll
    for (int r = 0; r < 8; ++r) {
      int m = mb + r;
      hB[m][n] = (_Float16)(tanh_fast(acc[r] + fb1[ctype * FH + n]) + (float)hA[m][n]);
    }
  }
  __syncthreads();

  // GEMM2: hA = tanh(hB @ W2 + b2) + hB
#pragma unroll
  for (int ni = 0; ni < 4; ++ni) {
    int nt = wave + 4 * ni;
    if (nt >= 15) continue;
    v8f acc = {0.f,0.f,0.f,0.f,0.f,0.f,0.f,0.f};
#pragma unroll
    for (int kt = 0; kt < 8; ++kt) {
      v16h afr = frag_a(&hB[0][kt * 32], 256, lane);
      v16h bfr = frag_b(fW2T + (size_t)(nt * 16) * 256 + kt * 32, 256, lane);
      acc = WMMA(afr, bfr, acc);
    }
    int n  = nt * 16 + (lane & 15);
    int mb = (lane & 16) ? 8 : 0;
#pragma unroll
    for (int r = 0; r < 8; ++r) {
      int m = mb + r;
      hA[m][n] = (_Float16)(tanh_fast(acc[r] + fb2[ctype * FH + n]) + (float)hB[m][n]);
    }
  }
  __syncthreads();

  // final: out = hA @ W3 + b3
  if (t < 16) {
    int m = t;
    float v = fb3[ctype];
    for (int k = 0; k < FH; ++k) v += (float)hA[m][k] * fW3[ctype * FH + k];
    int row = mtile * 16 + m;
    int bb = row >> 8, nn = row & 255;
    out[bb * 512 + ctype * 256 + nn] = v;
  }
}

extern "C" void kernel_launch(void* const* d_in, const int* in_sizes, int n_in,
                              void* d_out, int out_size, void* d_ws, size_t ws_size,
                              hipStream_t stream) {
  const float* Ri  = (const float*)d_in[0];
  const float* eW0 = (const float*)d_in[1];
  const float* eb0 = (const float*)d_in[2];
  const float* eW1 = (const float*)d_in[3];
  const float* eb1 = (const float*)d_in[4];
  const float* eW2 = (const float*)d_in[5];
  const float* eb2 = (const float*)d_in[6];
  const float* fW0 = (const float*)d_in[7];
  const float* fb0 = (const float*)d_in[8];
  const float* fW1 = (const float*)d_in[9];
  const float* fb1 = (const float*)d_in[10];
  const float* fW2 = (const float*)d_in[11];
  const float* fb2 = (const float*)d_in[12];
  const float* fW3 = (const float*)d_in[13];
  const float* fb3 = (const float*)d_in[14];
  _Float16* ws = (_Float16*)d_ws;
  float* out = (float*)d_out;

  prep_kernel<<<512, 256, 0, stream>>>(eW1, eW2, fW0, fW1, fW2, ws);
  embed_kernel<<<2048, 128, 0, stream>>>(Ri, eW0, eb0, eb1, eb2, ws, ws + OFF_DR);
  fit_kernel<<<128, 128, 0, stream>>>(ws, fb0, fb1, fb2, fW3, fb3, out);
}
